// DrmmTKSRanker_4105988735732
// MI455X (gfx1250) — compile-verified, matrix operand order checked by reading, not run on hardware
//
#include <hip/hip_runtime.h>
#include <hip/hip_bf16.h>

typedef __attribute__((ext_vector_type(16))) _Float16 v16h;
typedef __attribute__((ext_vector_type(8)))  _Float16 v8h;
typedef __attribute__((ext_vector_type(8)))  float    v8f;

#define QL    64
#define DL    4096
#define DIM   300
#define KP    320          // DIM padded to multiple of 32 (10 WMMA K-steps)
#define NT    64           // docs per tile
#define NTILES (DL / NT)
#define TOPK  20

__launch_bounds__(256, 1)
__global__ void drmm_tks_fused(const int*   __restrict__ qtok,
                               const int*   __restrict__ dtok,
                               const float* __restrict__ emb,
                               const float* __restrict__ attn_w,
                               const float* __restrict__ attn_b,
                               const float* __restrict__ lin_w,
                               const float* __restrict__ lin_b,
                               const float* __restrict__ out_w,
                               const float* __restrict__ out_b,
                               float*       __restrict__ out)
{
    // ~104 KB LDS: one workgroup per WGP (320 KB available on CDNA5)
    __shared__ __align__(16) _Float16 sA[QL * KP];   // normalized query, f16 (40 KB)
    __shared__ __align__(16) _Float16 sB[NT * KP];   // raw doc tile,   f16 (40 KB)
    __shared__ float sScore[QL * NT];                // 16 KB score tile
    __shared__ float sTopk[QL * TOPK];               // running top-20 per row
    __shared__ float sInvDoc[NT];
    __shared__ float sProb[QL];
    __shared__ float sRed[QL];

    const int b    = blockIdx.x;
    const int tid  = threadIdx.x;
    const int wave = tid >> 5;       // 8 waves (wave32)
    const int lane = tid & 31;

    // ---- Phase 0: gather queries, fp32-normalize -> f16, attn scores ----
    for (int m = wave; m < QL; m += 8) {
        const int tok = qtok[b * QL + m] + 1;
        const float* row = emb + (long)tok * DIM;
        float vals[10];
        float ss = 0.f, as = 0.f;
        #pragma unroll
        for (int i = 0; i < 10; ++i) {
            const int k = lane + i * 32;
            const float v = (k < DIM) ? row[k] : 0.f;
            const float w = (k < DIM) ? attn_w[k] : 0.f;
            vals[i] = v; ss += v * v; as += v * w;
        }
        #pragma unroll
        for (int off = 16; off > 0; off >>= 1) {
            ss += __shfl_xor(ss, off, 32);
            as += __shfl_xor(as, off, 32);
        }
        const float inv = 1.f / fmaxf(sqrtf(ss), 1e-12f);
        #pragma unroll
        for (int i = 0; i < 10; ++i)
            sA[m * KP + lane + i * 32] = (_Float16)(vals[i] * inv);
        if (lane == 0) sScore[m] = as + attn_b[0];   // stash attn logits
    }
    __syncthreads();

    if (tid == 0) {                                  // softmax over 64 logits
        float mx = -3.0e38f;
        for (int m = 0; m < QL; ++m) mx = fmaxf(mx, sScore[m]);
        float s = 0.f;
        for (int m = 0; m < QL; ++m) { float e = __expf(sScore[m] - mx); sProb[m] = e; s += e; }
        const float inv = 1.f / s;
        for (int m = 0; m < QL; ++m) sProb[m] *= inv;
    }
    if (tid < QL) {
        #pragma unroll
        for (int k = 0; k < TOPK; ++k) sTopk[tid * TOPK + k] = -3.0e38f;
    }
    __syncthreads();

    // ---- Main loop: 64-doc tiles, WMMA scores, fused top-k ----
    for (int tile = 0; tile < NTILES; ++tile) {
        // gather doc tile: raw f16 into sB, fp32 inverse norm into sInvDoc
        for (int dl = wave; dl < NT; dl += 8) {
            const int tok = dtok[b * DL + tile * NT + dl] + 1;
            const float* row = emb + (long)tok * DIM;
            float ss = 0.f;
            #pragma unroll
            for (int i = 0; i < 10; ++i) {
                const int k = lane + i * 32;
                const float v = (k < DIM) ? row[k] : 0.f;
                ss += v * v;
                sB[dl * KP + k] = (_Float16)v;
            }
            #pragma unroll
            for (int off = 16; off > 0; off >>= 1) ss += __shfl_xor(ss, off, 32);
            if (lane == 0) sInvDoc[dl] = 1.f / fmaxf(sqrtf(ss), 1e-12f);
        }
        __syncthreads();

        // wave -> (m-tile, 2 n-tiles). Fragment layout per CDNA5 16-bit A/B spec:
        // lanes 0-15: K chunks {0-7,16-23}; lanes 16-31: {8-15,24-31}; row = lane&15.
        const int mbase = (wave & 3) * 16;
        const int rlane = lane & 15;
        const int khalf = (lane < 16) ? 0 : 8;
        #pragma unroll
        for (int j = 0; j < 2; ++j) {
            const int nbase = (wave >> 2) * 32 + j * 16;
            v8f c = {};
            #pragma unroll
            for (int ks = 0; ks < 10; ++ks) {
                const int k0 = ks * 32 + khalf;
                const v8h alo = *(const v8h*)&sA[(mbase + rlane) * KP + k0];
                const v8h ahi = *(const v8h*)&sA[(mbase + rlane) * KP + k0 + 16];
                const v8h blo = *(const v8h*)&sB[(nbase + rlane) * KP + k0];
                const v8h bhi = *(const v8h*)&sB[(nbase + rlane) * KP + k0 + 16];
                v16h a, bm;
                #pragma unroll
                for (int i = 0; i < 8; ++i) {
                    a[i] = alo[i];  a[8 + i]  = ahi[i];
                    bm[i] = blo[i]; bm[8 + i] = bhi[i];
                }
                c = __builtin_amdgcn_wmma_f32_16x16x32_f16(
                        false, a, false, bm, (short)0, c, false, false);
            }
            // C layout: lanes 0-15 -> N=lane, M=vgpr; lanes 16-31 -> M=vgpr+8
            const int n    = nbase + rlane;
            const float iv = sInvDoc[n];
            const int mrow = mbase + ((lane < 16) ? 0 : 8);
            #pragma unroll
            for (int i = 0; i < 8; ++i)
                sScore[(mrow + i) * NT + n] = c[i] * iv;
        }
        __syncthreads();

        // fold tile into running sorted top-20 (one thread per query row)
        if (tid < QL) {
            float* tk = &sTopk[tid * TOPK];
            const float* srow = &sScore[tid * NT];
            for (int j = 0; j < NT; ++j) {
                const float v = srow[j];
                if (v > tk[TOPK - 1]) {
                    int p = TOPK - 1;
                    while (p > 0 && tk[p - 1] < v) { tk[p] = tk[p - 1]; --p; }
                    tk[p] = v;
                }
            }
        }
        __syncthreads();
    }

    // ---- Epilogue: linear(topk) * softmax, reduce, scalar affine ----
    if (tid < QL) {
        float dense = lin_b[0];
        #pragma unroll
        for (int k = 0; k < TOPK; ++k) dense += sTopk[tid * TOPK + k] * lin_w[k];
        sRed[tid] = dense * sProb[tid];
    }
    __syncthreads();
    if (tid == 0) {
        float s = 0.f;
        for (int m = 0; m < QL; ++m) s += sRed[m];
        out[b] = s * out_w[0] + out_b[0];
    }
}

extern "C" void kernel_launch(void* const* d_in, const int* in_sizes, int n_in,
                              void* d_out, int out_size, void* d_ws, size_t ws_size,
                              hipStream_t stream) {
    const int*   qtok   = (const int*)  d_in[0];   // [B, 64]
    const int*   dtok   = (const int*)  d_in[1];   // [B, 4096]
    const float* emb    = (const float*)d_in[2];   // [100000, 300]
    const float* attn_w = (const float*)d_in[3];   // [300]
    const float* attn_b = (const float*)d_in[4];   // [1]
    const float* lin_w  = (const float*)d_in[5];   // [20]
    const float* lin_b  = (const float*)d_in[6];   // [1]
    const float* out_w  = (const float*)d_in[7];   // [1]
    const float* out_b  = (const float*)d_in[8];   // [1]
    float*       out    = (float*)d_out;           // [B, 1]

    const int B = in_sizes[0] / QL;                // 128
    drmm_tks_fused<<<dim3(B), dim3(256), 0, stream>>>(
        qtok, dtok, emb, attn_w, attn_b, lin_w, lin_b, out_w, out_b, out);
}
